// ETSFormer_84885733638561
// MI455X (gfx1250) — matrix operand order repaired and checked by reading, not compiled, not run on hardware
//
#include <hip/hip_runtime.h>
#include <math.h>

typedef _Float16 half_t;
typedef __attribute__((ext_vector_type(16))) _Float16 v16h;
typedef __attribute__((ext_vector_type(8)))  float    v8f;

#define B_   32
#define N_   1024
#define D_   512
#define TF_  7
#define DFF_ 2048
#define HOR_ 96
#define M_   (B_ * N_)

enum { MODE_PLAIN = 0, MODE_SIGMOID_H16 = 1, MODE_SUB_AUX = 2 };

__device__ __forceinline__ float sigmoidf_(float x) { return 1.f / (1.f + __expf(-x)); }

// generic pointer -> raw LDS byte offset (low 32 bits of LDS aperture address)
__device__ __forceinline__ unsigned lds_off_(const void* p) {
  return (unsigned)(unsigned long long)p;
}

// ---------------------------------------------------------------------------
// Weight pre-pass: f32 [K][N] -> f16 [N][K]  (done once; weights re-read 256x)
// ---------------------------------------------------------------------------
__global__ __launch_bounds__(256) void transpose_to_h(const float* __restrict__ W,
                                                      half_t* __restrict__ Wt,
                                                      int K, int N)
{
  long idx = (long)blockIdx.x * 256 + threadIdx.x;
  if (idx >= (long)K * N) return;
  int k = (int)(idx / N);
  int n = (int)(idx % N);
  Wt[(long)n * K + k] = (half_t)W[idx];
}

// ---------------------------------------------------------------------------
// Front conv: z[b,t,d] = conv_b[d] + sum_{c<7,k<3} x[b,t+k-1,c] * w[d,c,k]
// ---------------------------------------------------------------------------
__global__ __launch_bounds__(256) void conv_front(const float* __restrict__ x,
                                                  const float* __restrict__ w,
                                                  const float* __restrict__ bias,
                                                  float* __restrict__ z)
{
  long idx = (long)blockIdx.x * 256 + threadIdx.x;
  if (idx >= (long)M_ * D_) return;
  int dcol = (int)(idx & (D_ - 1));
  int t    = (int)((idx >> 9) & (N_ - 1));
  int b    = (int)(idx >> 19);
  float acc = bias[dcol];
  #pragma unroll
  for (int k = 0; k < 3; k++) {
    int tt = t + k - 1;
    if (tt < 0 || tt >= N_) continue;
    const float* xp = x + ((long)b * N_ + tt) * TF_;
    const float* wp = w + (long)dcol * (TF_ * 3) + k;
    #pragma unroll
    for (int c = 0; c < TF_; c++) acc += xp[c] * wp[c * 3];
  }
  z[idx] = acc;
}

// ---------------------------------------------------------------------------
// In-LDS 1024-point radix-2 FFT (input must be pre-bit-reversed in re/im).
// ---------------------------------------------------------------------------
__device__ __forceinline__ void fft1024(float* re, float* im, int tid, float sign)
{
  #pragma unroll
  for (int st = 1; st <= 10; st++) {
    const int half = 1 << (st - 1);
    const float ang0 = sign * (6.2831853071795864f / (float)(1 << st));
    for (int j = tid; j < 512; j += 256) {
      int pos = j & (half - 1);
      int grp = j >> (st - 1);
      int i0  = (grp << st) + pos;
      int i1  = i0 + half;
      float s, c;
      __sincosf(ang0 * (float)pos, &s, &c);
      float br = re[i1] * c - im[i1] * s;
      float bi = re[i1] * s + im[i1] * c;
      float ar = re[i0], ai = im[i0];
      re[i0] = ar + br; im[i0] = ai + bi;
      re[i1] = ar - br; im[i1] = ai - bi;
    }
    __syncthreads();
  }
}

__device__ __forceinline__ int rev10(int i) { return (int)(__brev((unsigned)i) >> 22); }

// freq_attention: per (b,d) column; FFT -> top-4 amp threshold -> filter ->
// iFFT; writes lp, z -= lp in place, and f16 shadow of new z for the GEMM.
__global__ __launch_bounds__(256) void freq_att_topk(float* __restrict__ z,
                                                     float* __restrict__ lp,
                                                     half_t* __restrict__ z_h)
{
  __shared__ float re[1024];
  __shared__ float im[1024];
  __shared__ float amp[513];
  __shared__ float rmax[256];
  __shared__ int   ridx[256];
  __shared__ float kth_s;

  const int tid = threadIdx.x;
  const int b   = blockIdx.x >> 9;
  const int d   = blockIdx.x & (D_ - 1);
  const long base = (long)b * N_ * D_ + d;

  for (int i = tid; i < 1024; i += 256) {
    float v = z[base + (long)i * D_];
    int j = rev10(i);
    re[j] = v; im[j] = 0.f;
  }
  __syncthreads();
  fft1024(re, im, tid, -1.0f);            // forward

  for (int f = tid; f < 513; f += 256)
    amp[f] = sqrtf(re[f] * re[f] + im[f] * im[f]);
  __syncthreads();

  // 4 extraction rounds -> kth = 4th largest amplitude
  for (int r = 0; r < 4; r++) {
    float mv = -1.f; int mi = 0;
    for (int f = tid; f < 513; f += 256)
      if (amp[f] > mv) { mv = amp[f]; mi = f; }
    rmax[tid] = mv; ridx[tid] = mi;
    __syncthreads();
    for (int s = 128; s > 0; s >>= 1) {
      if (tid < s && rmax[tid + s] > rmax[tid]) {
        rmax[tid] = rmax[tid + s]; ridx[tid] = ridx[tid + s];
      }
      __syncthreads();
    }
    if (tid == 0) {
      if (r == 3) kth_s = rmax[0];
      amp[ridx[0]] = -2.f;                // mark extracted (amp >= 0 always)
    }
    __syncthreads();
  }
  const float kth = kth_s;

  // exact Hermitian symmetry in upper half (matches irfft semantics)
  for (int f = tid + 513; f < 1024; f += 256) {
    re[f] = re[1024 - f];
    im[f] = -im[1024 - f];
  }
  __syncthreads();

  // filter: keep if amp >= kth (extracted entries are in top-4 => keep)
  for (int f = tid; f < 1024; f += 256) {
    int fm = (f <= 512) ? f : 1024 - f;
    float a = amp[fm];
    bool keep = (a >= kth) || (a < -1.f);
    if (!keep) { re[f] = 0.f; im[f] = 0.f; }
  }
  __syncthreads();

  // bit-reverse permute for the inverse pass
  for (int i = tid; i < 1024; i += 256) {
    int j = rev10(i);
    if (i < j) {
      float tr = re[i], ti = im[i];
      re[i] = re[j]; im[i] = im[j];
      re[j] = tr;    im[j] = ti;
    }
  }
  __syncthreads();
  fft1024(re, im, tid, +1.0f);            // inverse (unscaled)

  for (int i = tid; i < 1024; i += 256) {
    float v = re[i] * (1.0f / 1024.0f);
    long o = base + (long)i * D_;
    float nz = z[o] - v;
    lp[o]  = v;
    z[o]   = nz;
    z_h[o] = (half_t)nz;
  }
}

// ---------------------------------------------------------------------------
// WMMA GEMM: out[M,N] = f(A[M,K] @ Wt[N,K]^T + bias)
// A: f16 [M][K] row-major; Wt: f16 [N][K] (pre-transposed weights).
// Double-buffered LDS tiles filled by global_load_async_to_lds_b128 (ASYNCcnt)
// so the DMA of tile k+1 overlaps the WMMAs of tile k.
// ---------------------------------------------------------------------------
__global__ __launch_bounds__(256) void gemm_wmma(const half_t* __restrict__ A,
                                                 const half_t* __restrict__ Wt,
                                                 const float* __restrict__ bias,
                                                 void* __restrict__ outv,
                                                 float* __restrict__ aux,
                                                 int M, int N, int K, int mode)
{
  __shared__ half_t As[2][128][40];   // [buf][m][k], rows padded to 80 B
  __shared__ half_t Bs[2][128][40];   // [buf][n][k]

  const int tid  = threadIdx.x;
  const int lane = tid & 31;
  const int wid  = tid >> 5;
  const int wm   = wid & 3;           // 4 wave rows  (32 m each)
  const int wn   = wid >> 2;          // 2 wave cols  (64 n each)
  const long bm0 = (long)blockIdx.x * 128;
  const long bn0 = (long)blockIdx.y * 128;

  // per-thread 16-byte async copy slot: row r, k-chunk kc
  const int r  = tid >> 1;            // 0..127
  const int kc = (tid & 1) * 16;      // 0 or 16
  const half_t* a_src = A  + (bm0 + r) * (long)K + kc;
  const half_t* b_src = Wt + (bn0 + r) * (long)K + kc;

  v8f acc[2][4];
  #pragma unroll
  for (int a = 0; a < 2; a++)
    #pragma unroll
    for (int b = 0; b < 4; b++)
      #pragma unroll
      for (int q = 0; q < 8; q++) acc[a][b][q] = 0.f;

  const int nk = K >> 5;

  // stage tile kt into buffer bufi (one b128 async copy per thread per tile)
  auto stage = [&](int bufi, int k0) {
    unsigned da = lds_off_(&As[bufi][r][kc]);
    unsigned db = lds_off_(&Bs[bufi][r][kc]);
    const half_t* pa = a_src + k0;
    const half_t* pb = b_src + k0;
    asm volatile("global_load_async_to_lds_b128 %0, %1, off"
                 :: "v"(da), "v"(pa) : "memory");
    asm volatile("global_load_async_to_lds_b128 %0, %1, off"
                 :: "v"(db), "v"(pb) : "memory");
  };

  stage(0, 0);
  for (int kt = 0; kt < nk; kt++) {
    asm volatile("s_wait_asynccnt 0x0" ::: "memory");  // own copies landed
    __syncthreads();                                    // everyone's landed
    if (kt + 1 < nk) stage((kt + 1) & 1, (kt + 1) << 5);

    const half_t (*as)[40] = As[kt & 1];
    const half_t (*bs)[40] = Bs[kt & 1];

    // fragment loads per CDNA5 WMMA VGPR layout (wave32)
    v16h af[2], bf[4];
    const int kb  = (lane >> 4) * 8;    // A: K {0..7|16..23} / {8..15|24..31}
    const int kb2 = (lane >> 4) * 16;   // B: K {0..7|16..23} / {8..15|24..31} (uses kb)
    #pragma unroll
    for (int mt = 0; mt < 2; mt++) {
      int m = wm * 32 + mt * 16 + (lane & 15);
      #pragma unroll
      for (int j = 0; j < 8; j++) af[mt][j]     = as[m][kb + j];
      #pragma unroll
      for (int j = 0; j < 8; j++) af[mt][8 + j] = as[m][16 + kb + j];
    }
    #pragma unroll
    for (int nt = 0; nt < 4; nt++) {
      int nn = wn * 64 + nt * 16 + (lane & 15);
      #pragma unroll
      for (int j = 0; j < 8; j++) bf[nt][j]     = bs[nn][kb + j];
      #pragma unroll
      for (int j = 0; j < 8; j++) bf[nt][8 + j] = bs[nn][16 + kb + j];
    }
    #pragma unroll
    for (int mt = 0; mt < 2; mt++)
      #pragma unroll
      for (int nt = 0; nt < 4; nt++)
        acc[mt][nt] = __builtin_amdgcn_wmma_f32_16x16x32_f16(
            false, af[mt], false, bf[nt], (short)0, acc[mt][nt], false, false);
    __syncthreads();
  }

  // epilogue: C/D layout -> n = lane&15, m = q + 8*(lane>>4)
  #pragma unroll
  for (int mt = 0; mt < 2; mt++) {
    #pragma unroll
    for (int nt = 0; nt < 4; nt++) {
      long n = bn0 + wn * 64 + nt * 16 + (lane & 15);
      float bv = bias[n];
      #pragma unroll
      for (int q = 0; q < 8; q++) {
        long m = bm0 + wm * 32 + mt * 16 + q + 8 * (lane >> 4);
        long o = m * (long)N + n;
        float v = acc[mt][nt][q] + bv;
        if (mode == MODE_SIGMOID_H16) {
          ((half_t*)outv)[o] = (half_t)sigmoidf_(v);
        } else if (mode == MODE_SUB_AUX) {
          ((float*)outv)[o] = v;
          aux[o] -= v;
        } else {
          ((float*)outv)[o] = v;
        }
      }
    }
  }
}

// ---------------------------------------------------------------------------
// mhesa EMA: conv1d_fft(xd, a(1-a)^rev) + (1-a)^(t+1)*init is exactly
// o[t] = (1-a) o[t-1] + a (x[t]-x[t-1]),  o[-1]=x[-1]=init.
// Reads f32, writes the f16 operand for the output projection GEMM.
// ---------------------------------------------------------------------------
__global__ __launch_bounds__(256) void ema_scan(const float* __restrict__ buf,
                                                half_t* __restrict__ out_h,
                                                const float* __restrict__ init,
                                                const float* __restrict__ alpha)
{
  int tid = blockIdx.x * 256 + threadIdx.x;
  if (tid >= B_ * D_) return;
  int col = tid & (D_ - 1);
  int b   = tid >> 9;
  int h   = col >> 6;
  float al = sigmoidf_(alpha[h]);
  float om = 1.f - al;
  float prev = init[col];
  float o    = prev;
  long idx = (long)b * N_ * D_ + col;
  for (int t = 0; t < N_; t++, idx += D_) {
    float xv = buf[idx];
    o = om * o + al * (xv - prev);
    prev = xv;
    out_h[idx] = (half_t)o;
  }
}

// ---------------------------------------------------------------------------
// LayerNorm over D=512 (one wave per row); optional residual input and
// optional f16 shadow output (GEMM operand).
// ---------------------------------------------------------------------------
__global__ __launch_bounds__(256) void layernorm_rows(const float* __restrict__ src,
                                                      const float* __restrict__ res,
                                                      const float* __restrict__ g,
                                                      const float* __restrict__ bta,
                                                      float* __restrict__ out,
                                                      half_t* __restrict__ out_h)
{
  int wid = threadIdx.x >> 5, lane = threadIdx.x & 31;
  long row = (long)blockIdx.x * 8 + wid;
  const float* sp = src + row * D_;
  float v[16];
  float sum = 0.f;
  #pragma unroll
  for (int i = 0; i < 16; i++) {
    float t = sp[lane + i * 32];
    if (res) t += res[row * D_ + lane + i * 32];
    v[i] = t; sum += t;
  }
  #pragma unroll
  for (int off = 16; off; off >>= 1) sum += __shfl_xor(sum, off, 32);
  float mu = sum * (1.f / 512.f);
  float var = 0.f;
  #pragma unroll
  for (int i = 0; i < 16; i++) { float dl = v[i] - mu; var += dl * dl; }
  #pragma unroll
  for (int off = 16; off; off >>= 1) var += __shfl_xor(var, off, 32);
  float rs = rsqrtf(var * (1.f / 512.f) + 1e-5f);
  float* op = out + row * D_;
  #pragma unroll
  for (int i = 0; i < 16; i++) {
    int c = lane + i * 32;
    float w = (v[i] - mu) * rs * g[c] + bta[c];
    op[c] = w;
    if (out_h) out_h[row * D_ + c] = (half_t)w;
  }
}

// ---------------------------------------------------------------------------
// Small-N projection (N = TF = 7): one wave per row, shuffle-reduced dots.
// ---------------------------------------------------------------------------
__global__ __launch_bounds__(256) void rowdot7(const float* __restrict__ A,
                                               const float* __restrict__ W,
                                               const float* __restrict__ bias,
                                               float* __restrict__ out)
{
  int wid = threadIdx.x >> 5, lane = threadIdx.x & 31;
  long row = (long)blockIdx.x * 8 + wid;
  const float* ap = A + row * D_;
  float p[TF_] = {0.f, 0.f, 0.f, 0.f, 0.f, 0.f, 0.f};
  for (int k = lane; k < D_; k += 32) {
    float av = ap[k];
    const float* wp = W + (long)k * TF_;
    #pragma unroll
    for (int c = 0; c < TF_; c++) p[c] += av * wp[c];
  }
  #pragma unroll
  for (int c = 0; c < TF_; c++)
    #pragma unroll
    for (int off = 16; off; off >>= 1) p[c] += __shfl_xor(p[c], off, 32);
  if (lane == 0) {
    float* op = out + row * TF_;
    #pragma unroll
    for (int c = 0; c < TF_; c++) op[c] = p[c] + bias[c];
  }
}

// ---------------------------------------------------------------------------
// level_step: x[t] = pt[t] + gt[t];  pt = (1-a)pt + a(x-per);  gt = (1-a)gt+gr
// (exact EMA form of the two conv1d_fft calls).  In-place on x.
// ---------------------------------------------------------------------------
__global__ __launch_bounds__(256) void level_scan(float* __restrict__ xc,
                                                  const float* __restrict__ per,
                                                  const float* __restrict__ gr,
                                                  const float* __restrict__ lalpha)
{
  int t = threadIdx.x;
  if (t >= B_ * TF_) return;
  int b = t / TF_, c = t % TF_;
  float al = sigmoidf_(lalpha[0]);
  float om = 1.f - al;
  float pt = 0.f, gt = 0.f;
  long idx = (long)b * N_ * TF_ + c;
  for (int i = 0; i < N_; i++, idx += TF_) {
    float xv = xc[idx];
    pt = om * pt + al * (xv - per[idx]);
    gt = om * gt + gr[idx];
    xc[idx] = pt + gt;
  }
}

// ---------------------------------------------------------------------------
// Snapshot lp[:, :96, :] (Fourier extrapolation at start==L is the periodic
// extension => first H samples) and lg[:, -1, :] (growth dampening seed).
// ---------------------------------------------------------------------------
__global__ __launch_bounds__(256) void snapshot(const float* __restrict__ lp,
                                                const float* __restrict__ lg,
                                                float* __restrict__ lp_snap,
                                                float* __restrict__ lg_snap)
{
  int idx = blockIdx.x * 256 + threadIdx.x;
  const int NS = B_ * HOR_ * D_;
  if (idx < NS) {
    int d = idx & (D_ - 1);
    int t = (idx >> 9) % HOR_;
    int b = idx / (HOR_ * D_);
    lp_snap[idx] = lp[((long)b * N_ + t) * D_ + d];
  }
  if (idx < B_ * D_) {
    int d = idx & (D_ - 1);
    int b = idx >> 9;
    lg_snap[idx] = lg[((long)b * N_ + (N_ - 1)) * D_ + d];
  }
}

// ---------------------------------------------------------------------------
// Final: agg = sum_l periods + sum_l damp ; out = level + agg @ out_w + out_b
// ---------------------------------------------------------------------------
__global__ __launch_bounds__(256) void final_forecast(const float* __restrict__ lp0,
                                                      const float* __restrict__ lp1,
                                                      const float* __restrict__ lg0,
                                                      const float* __restrict__ lg1,
                                                      const float* __restrict__ xc,
                                                      const float* __restrict__ dampen,
                                                      const float* __restrict__ out_w,
                                                      const float* __restrict__ out_b,
                                                      float* __restrict__ out)
{
  __shared__ float sagg[D_];
  int tid = threadIdx.x;
  int b  = blockIdx.x / HOR_;
  int tq = blockIdx.x % HOR_;
  #pragma unroll
  for (int ii = 0; ii < 2; ii++) {
    int d = tid + ii * 256;
    int h = d >> 6;
    float df = sigmoidf_(dampen[h]);
    // S(tq) = sum_{p=1}^{tq+1} df^p  (cumsum of df^powers)
    float S = df * (1.f - __powf(df, (float)(tq + 1))) / (1.f - df);
    long si = ((long)b * HOR_ + tq) * D_ + d;
    float lgv = lg0[b * D_ + d] + lg1[b * D_ + d];
    sagg[d] = lp0[si] + lp1[si] + lgv * S;
  }
  __syncthreads();
  if (tid < TF_) {
    float acc = out_b[tid] + xc[((long)b * N_ + (N_ - 1)) * TF_ + tid];
    for (int d = 0; d < D_; d++) acc += sagg[d] * out_w[d * TF_ + tid];
    out[((long)b * HOR_ + tq) * TF_ + tid] = acc;
  }
}

// ---------------------------------------------------------------------------
extern "C" void kernel_launch(void* const* d_in, const int* in_sizes, int n_in,
                              void* d_out, int out_size, void* d_ws, size_t ws_size,
                              hipStream_t stream)
{
  const float* x         = (const float*)d_in[0];
  const float* conv_w    = (const float*)d_in[1];
  const float* conv_b    = (const float*)d_in[2];
  const float* mh_init   = (const float*)d_in[3];
  const float* mh_alpha  = (const float*)d_in[4];
  const float* mh_win    = (const float*)d_in[5];
  const float* mh_bin    = (const float*)d_in[6];
  const float* mh_wout   = (const float*)d_in[7];
  const float* mh_bout   = (const float*)d_in[8];
  const float* ff_pre_g  = (const float*)d_in[9];
  const float* ff_pre_b  = (const float*)d_in[10];
  const float* ff_w1     = (const float*)d_in[11];
  const float* ff_b1     = (const float*)d_in[12];
  const float* ff_w2     = (const float*)d_in[13];
  const float* ff_b2     = (const float*)d_in[14];
  const float* ff_post_g = (const float*)d_in[15];
  const float* ff_post_b = (const float*)d_in[16];
  const float* lvl_alpha = (const float*)d_in[17];
  const float* lvl_wg    = (const float*)d_in[18];
  const float* lvl_bg    = (const float*)d_in[19];
  const float* lvl_wp    = (const float*)d_in[20];
  const float* lvl_bp    = (const float*)d_in[21];
  const float* dampen    = (const float*)d_in[22];
  const float* out_w     = (const float*)d_in[23];
  const float* out_b     = (const float*)d_in[24];
  // d_in[25] = forecast_horizon (96, fixed at compile time)

  const long SZ = (long)M_ * D_ * sizeof(float);   // 67 MB f32 activations
  unsigned char* p = (unsigned char*)d_ws;
  auto alloc = [&](size_t bytes) -> void* {
    void* r = (void*)p;
    p += (bytes + 255) & ~(size_t)255;
    return r;
  };
  float*  z       = (float*)alloc(SZ);
  float*  tmp1    = (float*)alloc(SZ);
  float*  lp      = (float*)alloc(SZ);
  float*  lg      = (float*)alloc(SZ);
  half_t* z_h     = (half_t*)alloc((size_t)M_ * D_ * sizeof(half_t));
  half_t* tmp1_h  = (half_t*)alloc((size_t)M_ * D_ * sizeof(half_t));
  half_t* h1h     = (half_t*)alloc((size_t)M_ * DFF_ * sizeof(half_t));
  float*  per     = (float*)alloc((size_t)M_ * TF_ * sizeof(float));
  float*  gr      = (float*)alloc((size_t)M_ * TF_ * sizeof(float));
  float*  xc      = (float*)alloc((size_t)M_ * TF_ * sizeof(float));
  float*  lp_snap = (float*)alloc((size_t)2 * B_ * HOR_ * D_ * sizeof(float));
  float*  lg_snap = (float*)alloc((size_t)2 * B_ * D_ * sizeof(float));
  // pre-transposed f16 weights [N][K]
  half_t* win_t[2], *wout_t[2];
  win_t[0]  = (half_t*)alloc((size_t)D_ * D_ * sizeof(half_t));
  win_t[1]  = (half_t*)alloc((size_t)D_ * D_ * sizeof(half_t));
  wout_t[0] = (half_t*)alloc((size_t)D_ * D_ * sizeof(half_t));
  wout_t[1] = (half_t*)alloc((size_t)D_ * D_ * sizeof(half_t));
  half_t* w1_t = (half_t*)alloc((size_t)D_ * DFF_ * sizeof(half_t));  // [2048][512]
  half_t* w2_t = (half_t*)alloc((size_t)DFF_ * D_ * sizeof(half_t));  // [512][2048]

  // one-time weight prep (cheap: ~6 MB of writes)
  {
    int nsq = (D_ * D_ + 255) / 256;
    for (int i = 0; i < 2; i++) {
      transpose_to_h<<<nsq, 256, 0, stream>>>(mh_win  + (long)i * D_ * D_, win_t[i],  D_, D_);
      transpose_to_h<<<nsq, 256, 0, stream>>>(mh_wout + (long)i * D_ * D_, wout_t[i], D_, D_);
    }
    int nff = (D_ * DFF_ + 255) / 256;
    transpose_to_h<<<nff, 256, 0, stream>>>(ff_w1, w1_t, D_, DFF_);
    transpose_to_h<<<nff, 256, 0, stream>>>(ff_w2, w2_t, DFF_, D_);
  }

  // level state starts as the raw input series
  hipMemcpyAsync(xc, x, (size_t)M_ * TF_ * sizeof(float),
                 hipMemcpyDeviceToDevice, stream);

  conv_front<<<(M_ * D_ + 255) / 256, 256, 0, stream>>>(x, conv_w, conv_b, z);

  for (int i = 0; i < 2; i++) {
    // periodicity component (FFT top-k filter), fused z -= lp (+ f16 shadow)
    freq_att_topk<<<B_ * D_, 256, 0, stream>>>(z, lp, z_h);

    // mhesa: in-proj -> per-head EMA scan -> out-proj (fused z -= lg)
    dim3 gsq(M_ / 128, D_ / 128);
    gemm_wmma<<<gsq, 256, 0, stream>>>(z_h, win_t[i], mh_bin + i * D_,
                                       tmp1, nullptr, M_, D_, D_, MODE_PLAIN);
    ema_scan<<<(B_ * D_ + 255) / 256, 256, 0, stream>>>(tmp1, tmp1_h,
                                                        mh_init + i * D_,
                                                        mh_alpha + i * 8);
    gemm_wmma<<<gsq, 256, 0, stream>>>(tmp1_h, wout_t[i], mh_bout + i * D_,
                                       lg, z, M_, D_, D_, MODE_SUB_AUX);

    if (i == 0) {  // FeedForward block on all but last layer
      layernorm_rows<<<M_ / 8, 256, 0, stream>>>(z, nullptr, ff_pre_g, ff_pre_b,
                                                 tmp1, tmp1_h);
      dim3 gw(M_ / 128, DFF_ / 128);
      gemm_wmma<<<gw, 256, 0, stream>>>(tmp1_h, w1_t, ff_b1, h1h, nullptr,
                                        M_, DFF_, D_, MODE_SIGMOID_H16);
      dim3 gn(M_ / 128, D_ / 128);
      gemm_wmma<<<gn, 256, 0, stream>>>(h1h, w2_t, ff_b2, z, nullptr,
                                        M_, D_, DFF_, MODE_PLAIN);
      layernorm_rows<<<M_ / 8, 256, 0, stream>>>(z, tmp1, ff_post_g, ff_post_b,
                                                 z, nullptr);
    }

    // level update
    rowdot7<<<M_ / 8, 256, 0, stream>>>(lp, lvl_wp + (long)i * D_ * TF_,
                                        lvl_bp + i * TF_, per);
    rowdot7<<<M_ / 8, 256, 0, stream>>>(lg, lvl_wg + (long)i * D_ * TF_,
                                        lvl_bg + i * TF_, gr);
    level_scan<<<1, 256, 0, stream>>>(xc, per, gr, lvl_alpha + i);

    snapshot<<<(B_ * HOR_ * D_ + 255) / 256, 256, 0, stream>>>(
        lp, lg, lp_snap + (long)i * B_ * HOR_ * D_, lg_snap + (long)i * B_ * D_);
  }

  final_forecast<<<B_ * HOR_, 256, 0, stream>>>(
      lp_snap, lp_snap + (long)B_ * HOR_ * D_,
      lg_snap, lg_snap + (long)B_ * D_,
      xc, dampen, out_w, out_b, (float*)d_out);
}
